// MP_model_52793738002617
// MI455X (gfx1250) — matrix operand order repaired and checked by reading, not compiled, not run on hardware
//
#include <hip/hip_runtime.h>
#include <hip/hip_bf16.h>

// ---------------------------------------------------------------------------
// GNN (NNConv-style) forward for MI455X / gfx1250.
// Dominant cost: per-edge 3-layer MLP (8->256->256->256) over 300K edges x 3
// MP layers (~2.4e11 FLOPs) -> fused f16 WMMA GEMMs (v_wmma_f32_16x16x32_f16,
// f32 accum).  Weights are repacked once into B-fragment lane order (two
// contiguous 16B loads per fragment per lane).  Each wave processes TWO
// 16-edge tiles so every B fragment is reused for 2 WMMAs (halves L2 weight
// traffic).  Activations live in LDS (32KB/block vs 320KB/WGP).
// ---------------------------------------------------------------------------

#define NN 20000
#define EE 300000
#define GG 64

typedef __attribute__((ext_vector_type(16))) _Float16 v16h;
typedef __attribute__((ext_vector_type(8)))  _Float16 v8h;
typedef __attribute__((ext_vector_type(8)))  float    v8f;
typedef __attribute__((ext_vector_type(4)))  unsigned int u32x4;
typedef __attribute__((ext_vector_type(4)))  float    f32x4;

union ABFrag {
    v16h     v;
    u32x4    q[2];
    _Float16 e[16];
};

__device__ __forceinline__ v8f wmma16(const ABFrag& a, const ABFrag& b, v8f c) {
    return __builtin_amdgcn_wmma_f32_16x16x32_f16(
        /*neg_a=*/false, a.v, /*neg_b=*/false, b.v,
        /*c_mod=*/(short)0, c, /*reuse_a=*/false, /*reuse_b=*/false);
}

// ---------------------------------------------------------------------------
// Weight repack: f32 [K][256] row-major -> f16 WMMA-B fragments.
// Fragment (ks,nt) covers B[ks*32 .. +31][nt*16 .. +15].
// Lane L owns 16 contiguous halves: k = ks*32 + (L<16?0:16) + h, n = nt*16+(L&15).
// Layer0 (K=8) zero-padded to K=32.  Per MP layer: 16 + 128 + 128 fragments.
// ---------------------------------------------------------------------------
__global__ __launch_bounds__(32) void k_convert(
    const float* __restrict__ w00, const float* __restrict__ w01, const float* __restrict__ w02,
    const float* __restrict__ w10, const float* __restrict__ w11, const float* __restrict__ w12,
    const float* __restrict__ w20, const float* __restrict__ w21, const float* __restrict__ w22,
    _Float16* __restrict__ wz)
{
    int fid = blockIdx.x;           // 0 .. 3*272-1
    int mp  = fid / 272;
    int r   = fid % 272;
    int layer, frag;
    if (r < 16)       { layer = 0; frag = r;       }
    else if (r < 144) { layer = 1; frag = r - 16;  }
    else              { layer = 2; frag = r - 144; }
    const float* W;
    switch (mp * 3 + layer) {
        case 0: W = w00; break; case 1: W = w01; break; case 2: W = w02; break;
        case 3: W = w10; break; case 4: W = w11; break; case 5: W = w12; break;
        case 6: W = w20; break; case 7: W = w21; break; default: W = w22; break;
    }
    const int Kdim = (layer == 0) ? 8 : 256;
    const int ks = frag >> 4, nt = frag & 15;
    const int lane = threadIdx.x;
    _Float16* dst = wz + ((size_t)mp * 272 + r) * 512 + lane * 16;
    const int n  = nt * 16 + (lane & 15);
    const int k0 = ks * 32 + ((lane < 16) ? 0 : 16);
#pragma unroll
    for (int h = 0; h < 16; ++h) {
        int k = k0 + h;
        float v = (k < Kdim) ? W[k * 256 + n] : 0.0f;
        dst[h] = (_Float16)v;
    }
}

// ---------------------------------------------------------------------------
// One 256->256 MLP layer for TWO 16-edge tiles (wave32): each B fragment is
// loaded once and used for 2 WMMAs.  A-fragment layout (16-bit A 16x32):
// lanes 0-15: M=0..15, K = s*32+{0..15}; lanes 16-31: +16 halves (contiguous k, as in B repack).
// ---------------------------------------------------------------------------
__device__ __forceinline__ void mlp_layer_256x2(
    const _Float16* __restrict__ s0, const _Float16* __restrict__ s1,
    _Float16* __restrict__ d0, _Float16* __restrict__ d1,
    const _Float16* __restrict__ Wf, const float* __restrict__ bias,
    int lane, int nloc, int khw, int rbase)
{
    for (int nt = 0; nt < 16; ++nt) {
        v8f acc0 = {};
        v8f acc1 = {};
#pragma unroll
        for (int ks = 0; ks < 8; ++ks) {
            ABFrag a0, a1, b;
            const u32x4* bp = (const u32x4*)(Wf + ((ks * 16 + nt) * 32 + lane) * 16);
            b.q[0] = bp[0];
            b.q[1] = bp[1];
            const u32x4* a0p = (const u32x4*)(s0 + nloc * 256 + ks * 32 + khw);
            a0.q[0] = a0p[0];
            a0.q[1] = a0p[1];                             // +8 halves
            const u32x4* a1p = (const u32x4*)(s1 + nloc * 256 + ks * 32 + khw);
            a1.q[0] = a1p[0];
            a1.q[1] = a1p[1];
            acc0 = wmma16(a0, b, acc0);
            acc1 = wmma16(a1, b, acc1);
        }
        float bv = bias[nt * 16 + nloc];
#pragma unroll
        for (int r = 0; r < 8; ++r) {                     // D: row = r + rbase, col = nloc
            d0[(rbase + r) * 256 + nt * 16 + nloc] = (_Float16)fmaxf(acc0[r] + bv, 0.0f);
            d1[(rbase + r) * 256 + nt * 16 + nloc] = (_Float16)fmaxf(acc1[r] + bv, 0.0f);
        }
    }
}

// ---------------------------------------------------------------------------
// Fused edge kernel: 1 wave = 32 edges (two 16-edge WMMA tiles).
// edge_attr -> 3-layer MLP (WMMA) -> per-edge [1,16]x[16,16] contraction with
// h[src] -> f32 atomic scatter into agg (deg division deferred to node kernel).
// ---------------------------------------------------------------------------
__global__ __launch_bounds__(32) void k_edge(
    const float* __restrict__ ea, const int* __restrict__ ei,
    const float* __restrict__ hcur, const _Float16* __restrict__ wz,
    const float* __restrict__ b0, const float* __restrict__ b1,
    const float* __restrict__ b2, float* __restrict__ agg)
{
    __shared__ _Float16 sA[2][16 * 256];
    __shared__ _Float16 sB[2][16 * 256];

    const int  lane   = threadIdx.x;
    const int  nloc   = lane & 15;
    const int  e0     = blockIdx.x * 32;
    const bool hiHalf = lane >= 16;
    const int  khw    = hiHalf ? 16 : 0;
    const int  rbase  = hiHalf ? 8 : 0;

    // ---- layer 0: A = edge_attr tiles (16 x 8 each), K zero-padded to 32 ----
    ABFrag a0[2];
#pragma unroll
    for (int t = 0; t < 2; ++t) {
        const f32x4* p = (const f32x4*)(ea + (e0 + t * 16 + nloc) * 8);
        f32x4 x0 = p[0], x1 = p[1];
        _Float16 z = (_Float16)0.0f;
        a0[t].e[0] = hiHalf ? z : (_Float16)x0.x;
        a0[t].e[1] = hiHalf ? z : (_Float16)x0.y;
        a0[t].e[2] = hiHalf ? z : (_Float16)x0.z;
        a0[t].e[3] = hiHalf ? z : (_Float16)x0.w;
        a0[t].e[4] = hiHalf ? z : (_Float16)x1.x;
        a0[t].e[5] = hiHalf ? z : (_Float16)x1.y;
        a0[t].e[6] = hiHalf ? z : (_Float16)x1.z;
        a0[t].e[7] = hiHalf ? z : (_Float16)x1.w;
#pragma unroll
        for (int h = 8; h < 16; ++h) a0[t].e[h] = z;
    }
    for (int nt = 0; nt < 16; ++nt) {
        ABFrag b;
        const u32x4* bp = (const u32x4*)(wz + (nt * 32 + lane) * 16);
        b.q[0] = bp[0];
        b.q[1] = bp[1];
        v8f acc0 = {};
        v8f acc1 = {};
        acc0 = wmma16(a0[0], b, acc0);
        acc1 = wmma16(a0[1], b, acc1);
        float bv = b0[nt * 16 + nloc];
#pragma unroll
        for (int r = 0; r < 8; ++r) {
            sA[0][(rbase + r) * 256 + nt * 16 + nloc] = (_Float16)fmaxf(acc0[r] + bv, 0.0f);
            sA[1][(rbase + r) * 256 + nt * 16 + nloc] = (_Float16)fmaxf(acc1[r] + bv, 0.0f);
        }
    }
    __syncthreads();
    mlp_layer_256x2(sA[0], sA[1], sB[0], sB[1], wz + 16 * 512,  b1, lane, nloc, khw, rbase);
    __syncthreads();
    mlp_layer_256x2(sB[0], sB[1], sA[0], sA[1], wz + 144 * 512, b2, lane, nloc, khw, rbase);
    __syncthreads();

    // ---- per-edge contraction: m[o] = sum_i h[src][i] * w[e][i*16+o] ----
    // 2 lanes per edge; 8 contiguous halves per (i) -> single ds_load_b128.
    const int e  = lane >> 1;
    const int o8 = (lane & 1) * 8;
#pragma unroll
    for (int t = 0; t < 2; ++t) {
        const int eg = e0 + t * 16 + e;
        const int sn = ei[eg];        // edge_index row 0 = src
        const int dn = ei[EE + eg];   // edge_index row 1 = dst
        const f32x4* hp = (const f32x4*)(hcur + sn * 16);
        f32x4 h0 = hp[0], h1 = hp[1], h2 = hp[2], h3 = hp[3];
        float hs[16] = { h0.x, h0.y, h0.z, h0.w, h1.x, h1.y, h1.z, h1.w,
                         h2.x, h2.y, h2.z, h2.w, h3.x, h3.y, h3.z, h3.w };
        float m[8] = {0.f, 0.f, 0.f, 0.f, 0.f, 0.f, 0.f, 0.f};
        const _Float16* act = sA[t];
#pragma unroll
        for (int i = 0; i < 16; ++i) {
            float hv = hs[i];
            v8h wv = *(const v8h*)(act + e * 256 + i * 16 + o8);
#pragma unroll
            for (int j = 0; j < 8; ++j)
                m[j] += hv * (float)wv[j];
        }
#pragma unroll
        for (int j = 0; j < 8; ++j)
            atomicAdd(&agg[dn * 16 + o8 + j], m[j]);
    }
}

// ---------------------------------------------------------------------------
// Small helper kernels (all bandwidth-trivial).
// ---------------------------------------------------------------------------
__global__ void k_zero(float* __restrict__ p, int n) {
    int t = blockIdx.x * blockDim.x + threadIdx.x;
    if (t < n) p[t] = 0.0f;
}

__global__ void k_deg(const int* __restrict__ ei, float* __restrict__ deg) {
    int e = blockIdx.x * blockDim.x + threadIdx.x;
    if (e < EE) atomicAdd(&deg[ei[EE + e]], 1.0f);
}

__global__ void k_node(const float* __restrict__ agg, const float* __restrict__ deg,
                       const float* __restrict__ hcur, const float* __restrict__ Wr,
                       const float* __restrict__ cb, float* __restrict__ hnext)
{
    int t = blockIdx.x * blockDim.x + threadIdx.x;
    if (t >= NN * 16) return;
    int n = t >> 4, o = t & 15;
    float acc = cb[o];
#pragma unroll
    for (int k = 0; k < 16; ++k) acc += hcur[n * 16 + k] * Wr[k * 16 + o];
    hnext[t] = agg[t] / fmaxf(deg[n], 1.0f) + acc;
}

__global__ void k_pool(const float* __restrict__ h, const int* __restrict__ batch,
                       float* __restrict__ pool, float* __restrict__ cnt)
{
    int t = blockIdx.x * blockDim.x + threadIdx.x;
    if (t >= NN * 16) return;
    int n = t >> 4, f = t & 15;
    int g = batch[n];
    atomicAdd(&pool[g * 16 + f], h[t]);
    if (f == 0) atomicAdd(&cnt[g], 1.0f);
}

__global__ void k_fc(const float* __restrict__ pool, const float* __restrict__ cnt,
                     const float* __restrict__ w0, const float* __restrict__ w1,
                     const float* __restrict__ w2, const float* __restrict__ w3,
                     const float* __restrict__ w4,
                     const float* __restrict__ b0, const float* __restrict__ b1,
                     const float* __restrict__ b2, const float* __restrict__ b3,
                     const float* __restrict__ b4, float* __restrict__ out)
{
    int g = threadIdx.x;
    if (g >= GG) return;
    const float* Ws[5] = {w0, w1, w2, w3, w4};
    const float* Bs[5] = {b0, b1, b2, b3, b4};
    float v[16];
    float c = fmaxf(cnt[g], 1.0f);
    for (int f = 0; f < 16; ++f) v[f] = pool[g * 16 + f] / c;
    for (int j = 0; j < 5; ++j) {
        int fo = (j == 4) ? 4 : 16;
        float o[16];
        for (int oo = 0; oo < fo; ++oo) {
            float a = Bs[j][oo];
            for (int k = 0; k < 16; ++k) a += v[k] * Ws[j][k * fo + oo];
            o[oo] = (j < 4) ? fmaxf(a, 0.0f) : a;
        }
        for (int oo = 0; oo < 16; ++oo) v[oo] = (oo < fo) ? o[oo] : 0.0f;
    }
    for (int oo = 0; oo < 4; ++oo) out[g * 4 + oo] = v[oo];
}

// ---------------------------------------------------------------------------
extern "C" void kernel_launch(void* const* d_in, const int* in_sizes, int n_in,
                              void* d_out, int out_size, void* d_ws, size_t ws_size,
                              hipStream_t stream)
{
    (void)in_sizes; (void)n_in; (void)out_size; (void)ws_size;
    const float* x     = (const float*)d_in[0];
    const int*   ei    = (const int*)d_in[1];
    const float* ea    = (const float*)d_in[2];
    const int*   batch = (const int*)d_in[3];
    const float* cew[9]; for (int i = 0; i < 9; ++i) cew[i] = (const float*)d_in[4 + i];
    const float* ceb[9]; for (int i = 0; i < 9; ++i) ceb[i] = (const float*)d_in[13 + i];
    const float* root[3]; for (int i = 0; i < 3; ++i) root[i] = (const float*)d_in[22 + i];
    const float* cb[3];   for (int i = 0; i < 3; ++i) cb[i]   = (const float*)d_in[25 + i];
    const float* fw[5];   for (int i = 0; i < 5; ++i) fw[i]   = (const float*)d_in[28 + i];
    const float* fb[5];   for (int i = 0; i < 5; ++i) fb[i]   = (const float*)d_in[33 + i];
    float* out = (float*)d_out;

    // workspace layout
    char* ws = (char*)d_ws;
    size_t off = 0;
    _Float16* wz = (_Float16*)(ws + off);  off += (size_t)3 * 272 * 512 * sizeof(_Float16);
    off = (off + 255) & ~(size_t)255;
    float* deg  = (float*)(ws + off);  off += (size_t)NN * sizeof(float);
    float* pool = (float*)(ws + off);  off += (size_t)GG * 16 * sizeof(float);
    float* cnt  = (float*)(ws + off);  off += (size_t)GG * sizeof(float);
    float* agg  = (float*)(ws + off);  off += (size_t)NN * 16 * sizeof(float);
    float* hA   = (float*)(ws + off);  off += (size_t)NN * 16 * sizeof(float);
    float* hB   = (float*)(ws + off);  off += (size_t)NN * 16 * sizeof(float);

    // 1) repack edge-MLP weights into WMMA B-fragment layout (f16)
    k_convert<<<3 * 272, 32, 0, stream>>>(cew[0], cew[1], cew[2], cew[3], cew[4],
                                          cew[5], cew[6], cew[7], cew[8], wz);
    // 2) zero deg+pool+cnt (contiguous), count in-degrees
    const int zc = NN + GG * 16 + GG;
    k_zero<<<(zc + 255) / 256, 256, 0, stream>>>(deg, zc);
    k_deg<<<(EE + 255) / 256, 256, 0, stream>>>(ei, deg);

    // 3) three message-passing layers
    const float* hc = x;
    float* hn = hA;
    for (int i = 0; i < 3; ++i) {
        k_zero<<<(NN * 16 + 255) / 256, 256, 0, stream>>>(agg, NN * 16);
        k_edge<<<EE / 32, 32, 0, stream>>>(ea, ei, hc, wz + (size_t)i * 272 * 512,
                                           ceb[3 * i], ceb[3 * i + 1], ceb[3 * i + 2], agg);
        k_node<<<(NN * 16 + 255) / 256, 256, 0, stream>>>(agg, deg, hc, root[i], cb[i], hn);
        hc = hn;
        hn = (hn == hA) ? hB : hA;
    }

    // 4) global mean pool + FC head
    k_pool<<<(NN * 16 + 255) / 256, 256, 0, stream>>>(hc, batch, pool, cnt);
    k_fc<<<1, 64, 0, stream>>>(pool, cnt, fw[0], fw[1], fw[2], fw[3], fw[4],
                               fb[0], fb[1], fb[2], fb[3], fb[4], out);
}
